// LTVZeroPhaseFIRFilter_22041772163078
// MI455X (gfx1250) — matrix operand order repaired and checked by reading, not compiled, not run on hardware
//
#include <hip/hip_runtime.h>
#include <hip/hip_bf16.h>

typedef __attribute__((ext_vector_type(16))) _Float16 v16h;
typedef __attribute__((ext_vector_type(8)))  _Float16 v8h;
typedef __attribute__((ext_vector_type(8)))  float    v8f;
typedef __attribute__((ext_vector_type(4)))  float    v4f;

#define B_     2
#define T_     131072
#define HOP    256
#define NBINS  257
#define NFFT   512
#define NFR    512        // frames present in log_mag
#define NFRO   511        // frames actually produced
#define KPAD   288        // 257 bins padded to 9 chunks of 32
#define HSTR   1024       // kernel-row stride (halfs); tap 0 lives at offset 240
#define HTAP0  240
#define XSTR   131584     // per-(shift,batch) signal row stride (halfs), padded coords
#define XPADL  255        // jnp.pad left amount: padded[i] = ex[i-255]
#define PI_F   3.14159265358979323846f

// ---- workspace layout (bytes) ----
#define EOFF_B   0                        // Ef16  [1024][288]  exp(log_mag) in f16
#define ESZ_B    (1024 * KPAD * 2)
#define MOFF_B   (EOFF_B + ESZ_B)         // Mt    [512][288]   basis, transposed (col-major in n)
#define MSZ_B    (512 * KPAD * 2)
#define HOFFS_B  (MOFF_B + MSZ_B)         // Hker  [1024][1024] per-frame FIR taps, zero margins
#define HSZ_B    (1024 * HSTR * 2)
#define ZERO_B   (HOFFS_B + HSZ_B)        // everything before here is pre-zeroed
#define XDUP_B   ZERO_B                   // Xdup  [16][2][XSTR] pre-shifted f16 signal
#define XDSZ_B   (16 * B_ * XSTR * 2)
#define WS_B     (XDUP_B + XDSZ_B)        // ~11.4 MB total

// ---------------------------------------------------------------------------
__global__ void k_zero_ws(unsigned int* __restrict__ ws, int nwords) {
    int i = blockIdx.x * blockDim.x + threadIdx.x;
    if (i < nwords) ws[i] = 0u;
}

// Ef16[r][k] = (f16) exp(log_mag[r, k]),  r = b*512 + frame
__global__ void k_fill_E(const float* __restrict__ lm, _Float16* __restrict__ E) {
    int i = blockIdx.x * blockDim.x + threadIdx.x;     // over 1024*KPAD
    if (i >= 1024 * KPAD) return;
    int r = i / KPAD, k = i % KPAD;
    if (k < NBINS) E[i] = (_Float16)__expf(lm[r * NBINS + k]);
}

// Mt[n][k] = hann(n)/512 * w_k * (-1)^k * cos(pi*k*n/256)   (irfft+fftshift+hann folded)
__global__ void k_fill_Mt(_Float16* __restrict__ Mt) {
    int i = blockIdx.x * blockDim.x + threadIdx.x;     // over 512*KPAD
    if (i >= 512 * KPAD) return;
    int n = i / KPAD, k = i % KPAD;
    if (k >= NBINS) return;
    float hann = 0.5f * (1.0f - __cosf((float)n * (PI_F / 256.0f)));
    float w    = (k == 0 || k == NFFT / 2) ? 1.0f : 2.0f;
    float s    = (k & 1) ? -1.0f : 1.0f;
    int   rr   = (k * n) & (NFFT - 1);                 // exact angle reduction
    float c    = __cosf((float)rr * (PI_F / 256.0f));
    Mt[i] = (_Float16)(hann * (1.0f / (float)NFFT) * w * s * c);
}

// Xdup[p][b][q] = padded[b][q+p] = (255 <= q+p < 255+T) ? ex[b][q+p-255] : 0
// 16 shifted copies so the Hankel A-fragment loads are aligned v8h per lane.
__global__ void k_fill_xdup(const float* __restrict__ ex, _Float16* __restrict__ Xd) {
    long i = (long)blockIdx.x * blockDim.x + threadIdx.x;   // over 16*B_*XSTR
    if (i >= (long)16 * B_ * XSTR) return;
    int q   = (int)(i % XSTR);
    int pb  = (int)(i / XSTR);                              // p*B_ + b
    int b   = pb & (B_ - 1);
    int p   = pb >> 1;
    int src = q + p - XPADL;                                // ex index
    _Float16 v = (_Float16)0.0f;
    if (src >= 0 && src < T_) v = (_Float16)ex[(long)b * T_ + src];
    Xd[i] = v;
}

// ---------------------------------------------------------------------------
// Stage 1: Hker[r][HTAP0 + n] = sum_k Ef16[r][k] * Mt[n][k]  (1024x512x288 GEMM)
// One wave per 16x16 tile; 2048 tiles; ISA-exact fragment layouts (wmma.md 7.12.2).
__global__ void k_gemm_dct(const _Float16* __restrict__ E, const _Float16* __restrict__ Mt,
                           _Float16* __restrict__ H) {
    int wave = threadIdx.x >> 5, lane = threadIdx.x & 31;
    int tile = blockIdx.x * 8 + wave;                  // [0, 2048)
    int tm = tile >> 5, tn = tile & 31;                // 64 x 32 tiles
    int m = lane & 15, hw = lane >> 4;
    int rowA = tm * 16 + m;
    int colB = tn * 16 + m;                            // B-fragment column = lane%16
    v8f acc = {};
    #pragma unroll
    for (int c = 0; c < KPAD / 32; ++c) {
        // A: lane holds row m, K = {Kb..Kb+7, Kb+16..Kb+23}, Kb = 32c + 8*hw
        const v8h* pa = (const v8h*)(E + (long)rowA * KPAD + c * 32 + hw * 8);
        v8h a0 = pa[0], a1 = pa[2];                    // +16 halfs = +2 v8h
        // B: lane holds col n, K = {Kb..Kb+7, Kb+16..Kb+23}, Kb = 32c + 8*hw (Mt is col-major in n)
        const v8h* pb = (const v8h*)(Mt + (long)colB * KPAD + c * 32 + hw * 8);
        v8h b0 = pb[0], b1 = pb[2];
        v16h av, bv;
        #pragma unroll
        for (int t = 0; t < 8; ++t) { av[t] = a0[t]; av[t + 8] = a1[t];
                                      bv[t] = b0[t]; bv[t + 8] = b1[t]; }
        acc = __builtin_amdgcn_wmma_f32_16x16x32_f16(false, av, false, bv,
                                                     (short)0, acc, false, false);
    }
    // D: lane holds col n = lane%16, rows m = v + 8*hw. Write taps into padded row.
    int tap = tn * 16 + m;
    #pragma unroll
    for (int v = 0; v < 8; ++v) {
        int row = tm * 16 + hw * 8 + v;
        H[(long)row * HSTR + HTAP0 + tap] = (_Float16)acc[v];
    }
}

// ---------------------------------------------------------------------------
// Stage 2: per-frame LTV FIR as 16x16x768 GEMM.
//   out[f, 16n+m] = sum_{k'} A[m,k'] B[k',n],
//   A[m,k'] = padded[f*256 + m + k'] = Xdup[m][f*256 + k']   (aligned v8h loads)
//   B[k',n] = h[k' - 16n]                                    (zero margins in H row)
__global__ void k_gemm_fir(const _Float16* __restrict__ Xd, const _Float16* __restrict__ H,
                           float* __restrict__ out) {
    int wave = threadIdx.x >> 5, lane = threadIdx.x & 31;
    int w = blockIdx.x * 8 + wave;                     // wave-uniform guard: EXEC stays full
    if (w >= B_ * NFRO) return;
    int b = w / NFRO, f = w % NFRO;
    int m = lane & 15, hw = lane >> 4;
    // A row for this lane: shift copy p = m, batch b; in-row base = f*256 (8-half aligned)
    const _Float16* xrow = Xd + ((long)m * B_ + b) * XSTR + (long)f * HOP;
    // shifted-kernel row pointer for column n = m; 16-half alignment guaranteed
    const _Float16* hrow = H + (long)(b * NFR + f) * HSTR + HTAP0 - 16 * m;
    __builtin_prefetch(hrow, 0, 1);                    // global_prefetch_b8
    __builtin_prefetch(xrow, 0, 1);
    v8f acc = {};
    #pragma unroll 4
    for (int c = 0; c < 24; ++c) {                     // K' = 768
        const v8h* pa = (const v8h*)(xrow + c * 32 + hw * 8);
        v8h a0 = pa[0], a1 = pa[2];                    // K = {Kb..Kb+7, Kb+16..Kb+23}
        const v8h* pb = (const v8h*)(hrow + c * 32 + hw * 8);
        v8h b0 = pb[0], b1 = pb[2];                    // K = {Kb..Kb+7, Kb+16..Kb+23}
        v16h av, bv;
        #pragma unroll
        for (int t = 0; t < 8; ++t) { av[t] = a0[t]; av[t + 8] = a1[t];
                                      bv[t] = b0[t]; bv[t + 8] = b1[t]; }
        acc = __builtin_amdgcn_wmma_f32_16x16x32_f16(false, av, false, bv,
                                                     (short)0, acc, false, false);
    }
    // D: col n = lane%16, rows m = 8*hw + v  ->  j = 16n + 8*hw + v (8 consecutive floats)
    float* obase = out + (long)b * (NFRO * HOP) + (long)f * HOP + 16 * m + 8 * hw;
    v4f lo = {acc[0], acc[1], acc[2], acc[3]};
    v4f hi = {acc[4], acc[5], acc[6], acc[7]};
    *(v4f*)obase       = lo;
    *(v4f*)(obase + 4) = hi;
}

// ---------------------------------------------------------------------------
extern "C" void kernel_launch(void* const* d_in, const int* in_sizes, int n_in,
                              void* d_out, int out_size, void* d_ws, size_t ws_size,
                              hipStream_t stream) {
    const float* ex = (const float*)d_in[0];           // (2, 131072) f32
    const float* lm = (const float*)d_in[1];           // (2, 512, 257) f32
    float* out = (float*)d_out;                        // (2, 130816) f32

    char* ws = (char*)d_ws;
    _Float16* E  = (_Float16*)(ws + EOFF_B);
    _Float16* Mt = (_Float16*)(ws + MOFF_B);
    _Float16* H  = (_Float16*)(ws + HOFFS_B);
    _Float16* Xd = (_Float16*)(ws + XDUP_B);

    int zwords = ZERO_B / 4;                           // zero E + Mt + H (margins matter)
    k_zero_ws<<<(zwords + 255) / 256, 256, 0, stream>>>((unsigned int*)ws, zwords);
    k_fill_E   <<<(1024 * KPAD + 255) / 256, 256, 0, stream>>>(lm, E);
    k_fill_Mt  <<<(512 * KPAD + 255) / 256, 256, 0, stream>>>(Mt);
    {
        long n = (long)16 * B_ * XSTR;
        k_fill_xdup<<<(int)((n + 255) / 256), 256, 0, stream>>>(ex, Xd);
    }
    k_gemm_dct<<<256, 256, 0, stream>>>(E, Mt, H);     // 2048 tiles, 8 waves/block
    k_gemm_fir<<<128, 256, 0, stream>>>(Xd, H, out);   // 1022 frame-waves
}
